// CnnMulti2GruUser_19911468385051
// MI455X (gfx1250) — matrix operand order repaired and checked by reading, not compiled, not run on hardware
//
#include <hip/hip_runtime.h>
#include <hip/hip_bf16.h>

typedef __attribute__((ext_vector_type(16))) _Float16 v16h;
typedef __attribute__((ext_vector_type(8)))  float    v8f;

#define EP 224   // E padded to multiple of 32 (200 -> 224)
#define TP 136   // T=128 plus zeroed pad rows for k-1 overreach

// ---------------- fragment loaders (wave32, contiguous k = 16*(lane/16) + element, same in A and B) ----

// A 16x32 f16 fragment from a row-major f16 buffer (ld = row stride in halves).
// lanes 0-15: M=lane, VGPR j -> K=2j,2j+1
// lanes 16-31: same with K += 16
__device__ __forceinline__ v16h load_a_frag(const _Float16* src, int ld, int lane) {
  int g = lane >> 4;
  const _Float16* row = src + (lane & 15) * ld;
  union { v16h v; unsigned u[8]; } f;
#pragma unroll
  for (int j = 0; j < 8; ++j) {
    int kb = 2 * j + 16 * g;
    f.u[j] = *(const unsigned*)(row + kb);  // pair (K, K+1), 4B aligned
  }
  return f.v;
}

// B 32x16 f16 fragment from pre-swizzled storage:
// tile layout = [tileIdx][lane][16 halves] contiguous (one 32-byte load)
__device__ __forceinline__ v16h load_b_frag(const _Float16* base, int tileIdx, int lane) {
  return *(const v16h*)(base + ((size_t)tileIdx * 32 + lane) * 16);
}

__device__ __forceinline__ v8f wmma_f16(v16h a, v16h b, v8f c) {
  return __builtin_amdgcn_wmma_f32_16x16x32_f16(false, a, false, b, (short)0, c, false, false);
}

__device__ __forceinline__ float selu_f(float x) {
  const float a = 1.6732632423543772f, s = 1.0507009873554805f;
  return s * (x > 0.f ? x : a * (__expf(x) - 1.f));
}
__device__ __forceinline__ float sigm_f(float x) { return 1.f / (1.f + __expf(-x)); }

// ---------------- weight repack kernels (run every launch; deterministic) ----

// conv_w[k] (NF=128, E=200, k) -> B[K = dt*EP + e][f], swizzled fragment order.
// region sizes (halves): k=2:57344, k=3:86016, k=4:114688, k=5:143360  (total 401408)
__global__ __launch_bounds__(256) void repack_conv_k(
    const float* __restrict__ w2, const float* __restrict__ w3,
    const float* __restrict__ w4, const float* __restrict__ w5,
    _Float16* __restrict__ dst) {
  int gid = blockIdx.x * 256 + threadIdx.x;
  if (gid >= 401408) return;
  int kk, base;
  if (gid < 57344)       { kk = 0; base = 0; }
  else if (gid < 143360) { kk = 1; base = 57344; }
  else if (gid < 258048) { kk = 2; base = 143360; }
  else                   { kk = 3; base = 258048; }
  const float* w = (kk == 0) ? w2 : (kk == 1) ? w3 : (kk == 2) ? w4 : w5;
  int k = kk + 2;
  int rel = gid - base;
  int hh = rel & 15;
  int lane = (rel >> 4) & 31;
  int tileIdx = rel >> 9;           // 512 halves per (kt,nt) tile
  int kt = tileIdx >> 3, nt = tileIdx & 7;      // nNt = 8 for NF=128
  int K = kt * 32 + (lane >> 4) * 16 + hh;
  int dt = K / EP, e = K % EP;
  int f = nt * 16 + (lane & 15);
  float v = (e < 200) ? w[(f * 200 + e) * k + dt] : 0.f;
  dst[gid] = (_Float16)v;
}

// s_gru wih/whh (768,256) -> B[c][gate], swizzled. 4 mats: fwd wih, fwd whh, bwd wih, bwd whh
__global__ __launch_bounds__(256) void repack_sgru_k(
    const float* __restrict__ wih_f, const float* __restrict__ whh_f,
    const float* __restrict__ wih_b, const float* __restrict__ whh_b,
    _Float16* __restrict__ dst) {
  int gid = blockIdx.x * 256 + threadIdx.x;
  if (gid >= 786432) return;
  int mi = gid / 196608;
  const float* w = (mi == 0) ? wih_f : (mi == 1) ? whh_f : (mi == 2) ? wih_b : whh_b;
  int rel = gid % 196608;
  int hh = rel & 15;
  int lane = (rel >> 4) & 31;
  int tileIdx = rel >> 9;           // nNt = 48, nKt = 8
  int kt = tileIdx / 48, nt = tileIdx % 48;
  int c = kt * 32 + (lane >> 4) * 16 + hh;
  int gcol = nt * 16 + (lane & 15);
  dst[gid] = (_Float16)w[gcol * 256 + c];
}

// ---------------- kernel 1: fused embed-gather + 4-ngram conv + maxpool + proj ----

__global__ __launch_bounds__(256) void sentence_encode_k(
    const int* __restrict__ tokens,        // [1920][128]
    const float* __restrict__ embed,       // [50000][200]
    const _Float16* __restrict__ convswz,  // swizzled conv weights
    const float* __restrict__ cb2, const float* __restrict__ cb3,
    const float* __restrict__ cb4, const float* __restrict__ cb5,
    const float* __restrict__ proj_w,      // [512][256]
    const float* __restrict__ proj_b,      // [256]
    _Float16* __restrict__ sv)             // [1920][256] f16
{
  __shared__ _Float16 smem[TP * EP];   // 60928 B: padded f16 sentence image
  __shared__ int ids[128];
  __shared__ int fmaxI[512];           // relu-max as int bits (nonneg floats)

  const int s = blockIdx.x;
  const int tid = threadIdx.x;

  { // zero padded image + init
    unsigned* z = (unsigned*)smem;
    for (int i = tid; i < TP * EP / 2; i += 256) z[i] = 0u;
    if (tid < 128) ids[tid] = tokens[(size_t)s * 128 + tid];
    for (int i = tid; i < 512; i += 256) fmaxI[i] = 0;
  }
  __syncthreads();

  // gather + f32->f16
  for (int i = tid; i < 128 * 200; i += 256) {
    int t = i / 200, e = i - t * 200;
    smem[t * EP + e] = (_Float16)embed[(size_t)ids[t] * 200 + e];
  }
  __syncthreads();

  const int lane = tid & 31, wid = tid >> 5;
  const int nKt[4] = {14, 21, 28, 35};            // k*EP/32
  const int wbase[4] = {0, 57344, 143360, 258048};
  const float* cbs[4] = {cb2, cb3, cb4, cb5};

  for (int kk = 0; kk < 4; ++kk) {
    const int k = kk + 2;
    const int tlim = 128 - k + 1;                 // valid conv positions
    const _Float16* wsw = convswz + wbase[kk];
    const float* cb = cbs[kk];
    const int kTiles = nKt[kk];
    for (int tile = wid; tile < 64; tile += 8) {  // 8 Mtiles x 8 Ntiles per wave set
      int mt = tile >> 3, nt = tile & 7;
      v8f c = {};
      for (int kt = 0; kt < kTiles; ++kt) {
        v16h a = load_a_frag(smem + mt * 16 * EP + kt * 32, EP, lane);
        v16h b = load_b_frag(wsw, kt * 8 + nt, lane);
        c = wmma_f16(a, b, c);
      }
      int n = lane & 15, g = lane >> 4;
      int f = nt * 16 + n;
      float bias = cb[f];
      float m = 0.f;
#pragma unroll
      for (int r = 0; r < 8; ++r) {
        int t = mt * 16 + r + 8 * g;              // C layout: M = r + 8*(lane>=16)
        float v = c[r] + bias;
        v = (t < tlim) ? fmaxf(v, 0.f) : 0.f;     // relu + mask tail positions
        m = fmaxf(m, v);
      }
      m = fmaxf(m, __shfl_xor(m, 16, 32));        // combine the two M-halves (wave32)
      if (g == 0) atomicMax(&fmaxI[kk * 128 + f], __float_as_int(m));
    }
  }
  __syncthreads();

  // projection: feats[512] @ proj_w[512,256] + proj_b
  float acc = proj_b[tid];
  for (int j = 0; j < 512; ++j)
    acc += __int_as_float(fmaxI[j]) * proj_w[j * 256 + tid];
  sv[(size_t)s * 256 + tid] = (_Float16)acc;
}

// ---------------- kernel 2: s_gru x-gates for all (p,t), both dirs (WMMA GEMM) ----

__global__ __launch_bounds__(256) void sgru_xgates_k(
    const _Float16* __restrict__ sv,        // [1920][256] f16
    const _Float16* __restrict__ sgruswz,   // 4 swizzled mats
    const float* __restrict__ bih_f, const float* __restrict__ bih_b,
    float* __restrict__ gx)                 // [2][1920][768] f32
{
  const int lane = threadIdx.x & 31, wid = threadIdx.x >> 5;
  int wg = blockIdx.x * 8 + wid;            // 0..11519
  int dir = wg / 5760;
  int rem = wg - dir * 5760;
  int mt = rem / 48, nt = rem - mt * 48;    // M=1920 (120 tiles), N=768 (48 tiles)
  const _Float16* bsw = sgruswz + (size_t)(dir * 2) * 196608;  // wih of this dir
  const float* bih = dir ? bih_b : bih_f;

  v8f c = {};
  for (int kt = 0; kt < 8; ++kt) {          // K = 256
    v16h a = load_a_frag(sv + (size_t)mt * 16 * 256 + kt * 32, 256, lane);
    v16h b = load_b_frag(bsw, kt * 48 + nt, lane);
    c = wmma_f16(a, b, c);
  }
  int n = lane & 15, g = lane >> 4;
  int col = nt * 16 + n;
  float bb = bih[col];
  float* gout = gx + (size_t)dir * 1920 * 768;
#pragma unroll
  for (int r = 0; r < 8; ++r) {
    int m = mt * 16 + r + 8 * g;
    gout[(size_t)m * 768 + col] = c[r] + bb;
  }
}

// ---------------- kernel 3: sequential s_gru (1 block per direction) ----

__global__ __launch_bounds__(256) void sgru_seq_k(
    const _Float16* __restrict__ sgruswz,
    const float* __restrict__ bhh_f, const float* __restrict__ bhh_b,
    const float* __restrict__ gx,           // [2][1920][768]
    float* __restrict__ doc)                // [2][64][256]
{
  const int dir = blockIdx.x;
  const _Float16* bsw = sgruswz + (size_t)(dir * 2 + 1) * 196608;  // whh of this dir
  const float* bhh = dir ? bhh_b : bhh_f;
  const float* gxd = gx + (size_t)dir * 1920 * 768;

  __shared__ _Float16 h16[64 * 256];   // 32 KB
  __shared__ _Float16 gh16[64 * 768];  // 96 KB

  for (int i = threadIdx.x; i < 64 * 256; i += 256) h16[i] = (_Float16)0.f;
  __syncthreads();

  const int lane = threadIdx.x & 31, wid = threadIdx.x >> 5;
  for (int t = 0; t < 30; ++t) {
    // pass A: gh = h @ whh^T  (M=64: 4 tiles, N=768: 48 tiles, K=256: 8 steps)
    for (int tile = wid; tile < 192; tile += 8) {
      int mt = tile / 48, nt = tile - mt * 48;
      v8f c = {};
      for (int kt = 0; kt < 8; ++kt) {
        v16h a = load_a_frag(h16 + mt * 16 * 256 + kt * 32, 256, lane);
        v16h b = load_b_frag(bsw, kt * 48 + nt, lane);
        c = wmma_f16(a, b, c);
      }
      int n = lane & 15, g = lane >> 4;
      int col = nt * 16 + n;
#pragma unroll
      for (int r = 0; r < 8; ++r) {
        int m = mt * 16 + r + 8 * g;
        gh16[m * 768 + col] = (_Float16)c[r];
      }
    }
    __syncthreads();
    // pass B: elementwise gate math, h update
    for (int idx = threadIdx.x; idx < 64 * 256; idx += 256) {
      int b = idx >> 8, j = idx & 255;
      int row = b * 30 + (dir ? (29 - t) : t);
      const float* gr = gxd + (size_t)row * 768;
      float xr = gr[j], xz = gr[256 + j], xn = gr[512 + j];
      float hr = (float)gh16[b * 768 + j]       + bhh[j];
      float hz = (float)gh16[b * 768 + 256 + j] + bhh[256 + j];
      float hn = (float)gh16[b * 768 + 512 + j] + bhh[512 + j];
      float hp = (float)h16[idx];
      float r = sigm_f(xr + hr);
      float z = sigm_f(xz + hz);
      float n = tanhf(xn + r * hn);
      float hv = (1.f - z) * n + z * hp;
      h16[idx] = (_Float16)hv;
      if (t == 29) doc[(size_t)dir * 64 * 256 + idx] = hv;
    }
    __syncthreads();
  }
}

// ---------------- kernel 4: tail (pb, r_stars, r_gru, p_stars) ----

__global__ __launch_bounds__(768) void tail_k(
    const float* __restrict__ doc2,          // [2][64][256]
    const float* __restrict__ user_feats,    // [64][20]
    const float* __restrict__ rfc_w1, const float* __restrict__ rfc_b1,
    const float* __restrict__ rfc_w2, const float* __restrict__ rfc_b2,
    const float* __restrict__ pfc_w1, const float* __restrict__ pfc_b1,
    const float* __restrict__ pfc_w2, const float* __restrict__ pfc_b2,
    const float* __restrict__ wih_f, const float* __restrict__ whh_f,
    const float* __restrict__ bih_f, const float* __restrict__ bhh_f,
    const float* __restrict__ wih_b, const float* __restrict__ whh_b,
    const float* __restrict__ bih_b, const float* __restrict__ bhh_b,
    float* __restrict__ out)                 // [65]: p_stars, r_stars[64]
{
  __shared__ float pb[64 * 276];
  __shared__ float rhid[64 * 128];
  __shared__ float g1[768], g2[768];
  __shared__ float h[256], hsum[256];
  __shared__ float ph[128];
  __shared__ float norms[64];
  const int tid = threadIdx.x;

  if (tid < 64) {
    float s = 0.f;
    for (int u = 0; u < 20; ++u) { float v = user_feats[tid * 20 + u]; s += v * v; }
    norms[tid] = fmaxf(sqrtf(s), 1e-12f);
  }
  if (tid < 256) hsum[tid] = 0.f;
  __syncthreads();

  for (int i = tid; i < 64 * 276; i += 768) {
    int p = i / 276, j = i - p * 276;
    pb[i] = (j < 256) ? (doc2[p * 256 + j] + doc2[64 * 256 + p * 256 + j])
                      : user_feats[p * 20 + (j - 256)] / norms[p];
  }
  __syncthreads();

  // r_stars = selu(pb @ rfc_w1 + b1) @ rfc_w2 + b2
  for (int i = tid; i < 64 * 128; i += 768) {
    int p = i >> 7, m = i & 127;
    float a = rfc_b1[m];
    for (int j = 0; j < 276; ++j) a += pb[p * 276 + j] * rfc_w1[j * 128 + m];
    rhid[i] = selu_f(a);
  }
  __syncthreads();
  if (tid < 64) {
    float a = rfc_b2[0];
    for (int m = 0; m < 128; ++m) a += rhid[tid * 128 + m] * rfc_w2[m];
    out[1 + tid] = a;
  }

  // r_gru over the 64 docs (batch 1, in=276, hid=256), both directions
  for (int dir = 0; dir < 2; ++dir) {
    const float* wih = dir ? wih_b : wih_f;
    const float* whh = dir ? whh_b : whh_f;
    const float* bih = dir ? bih_b : bih_f;
    const float* bhh = dir ? bhh_b : bhh_f;
    __syncthreads();
    if (tid < 256) h[tid] = 0.f;
    __syncthreads();
    for (int step = 0; step < 64; ++step) {
      int row = dir ? (63 - step) : step;
      {
        float a = bih[tid];
        const float* x = &pb[row * 276];
        for (int c2 = 0; c2 < 276; ++c2) a += x[c2] * wih[tid * 276 + c2];
        float b = bhh[tid];
        for (int c2 = 0; c2 < 256; ++c2) b += h[c2] * whh[tid * 256 + c2];
        g1[tid] = a; g2[tid] = b;
      }
      __syncthreads();
      if (tid < 256) {
        float r = sigm_f(g1[tid] + g2[tid]);
        float z = sigm_f(g1[256 + tid] + g2[256 + tid]);
        float n = tanhf(g1[512 + tid] + r * g2[512 + tid]);
        h[tid] = (1.f - z) * n + z * h[tid];
      }
      __syncthreads();
    }
    if (tid < 256) hsum[tid] += h[tid];
  }
  __syncthreads();

  // p_stars = selu(hn @ pfc_w1 + b1) @ pfc_w2 + b2
  if (tid < 128) {
    float a = pfc_b1[tid];
    for (int j = 0; j < 256; ++j) a += hsum[j] * pfc_w1[j * 128 + tid];
    ph[tid] = selu_f(a);
  }
  __syncthreads();
  if (tid == 0) {
    float a = pfc_b2[0];
    for (int m = 0; m < 128; ++m) a += ph[m] * pfc_w2[m];
    out[0] = a;
  }
}

// ---------------- launch ----

extern "C" void kernel_launch(void* const* d_in, const int* in_sizes, int n_in,
                              void* d_out, int out_size, void* d_ws, size_t ws_size,
                              hipStream_t stream) {
  (void)in_sizes; (void)n_in; (void)out_size; (void)ws_size;
  // top-level: inputs, sent_lengths, sent_counts, user_feats, then params leaves
  // (jax tree order: sorted dict keys, recursively)
  const int*   tokens     = (const int*)d_in[0];
  const float* user_feats = (const float*)d_in[3];
  const float* cb2 = (const float*)d_in[4];
  const float* cb3 = (const float*)d_in[5];
  const float* cb4 = (const float*)d_in[6];
  const float* cb5 = (const float*)d_in[7];
  const float* cw2 = (const float*)d_in[8];
  const float* cw3 = (const float*)d_in[9];
  const float* cw4 = (const float*)d_in[10];
  const float* cw5 = (const float*)d_in[11];
  const float* embed  = (const float*)d_in[12];
  const float* pfc_b1 = (const float*)d_in[13];
  const float* pfc_b2 = (const float*)d_in[14];
  const float* pfc_w1 = (const float*)d_in[15];
  const float* pfc_w2 = (const float*)d_in[16];
  const float* proj_b = (const float*)d_in[17];
  const float* proj_w = (const float*)d_in[18];
  const float* rg_b_bhh = (const float*)d_in[19];
  const float* rg_b_bih = (const float*)d_in[20];
  const float* rg_b_whh = (const float*)d_in[21];
  const float* rg_b_wih = (const float*)d_in[22];
  const float* rg_f_bhh = (const float*)d_in[23];
  const float* rg_f_bih = (const float*)d_in[24];
  const float* rg_f_whh = (const float*)d_in[25];
  const float* rg_f_wih = (const float*)d_in[26];
  const float* rfc_b1 = (const float*)d_in[27];
  const float* rfc_b2 = (const float*)d_in[28];
  const float* rfc_w1 = (const float*)d_in[29];
  const float* rfc_w2 = (const float*)d_in[30];
  const float* sg_b_bhh = (const float*)d_in[31];
  const float* sg_b_bih = (const float*)d_in[32];
  const float* sg_b_whh = (const float*)d_in[33];
  const float* sg_b_wih = (const float*)d_in[34];
  const float* sg_f_bhh = (const float*)d_in[35];
  const float* sg_f_bih = (const float*)d_in[36];
  const float* sg_f_whh = (const float*)d_in[37];
  const float* sg_f_wih = (const float*)d_in[38];

  char* ws = (char*)d_ws;
  _Float16* convswz = (_Float16*)(ws);                 //   802,816 B
  _Float16* sgruswz = (_Float16*)(ws + 802816);        // 1,572,864 B
  _Float16* sv      = (_Float16*)(ws + 2375680);       //   983,040 B
  float*    gx      = (float*)  (ws + 3358720);        // 11,796,480 B
  float*    doc     = (float*)  (ws + 15155200);       //   131,072 B

  repack_conv_k<<<1568, 256, 0, stream>>>(cw2, cw3, cw4, cw5, convswz);
  repack_sgru_k<<<3072, 256, 0, stream>>>(sg_f_wih, sg_f_whh, sg_b_wih, sg_b_whh, sgruswz);
  sentence_encode_k<<<1920, 256, 0, stream>>>(tokens, embed, convswz,
                                              cb2, cb3, cb4, cb5, proj_w, proj_b, sv);
  sgru_xgates_k<<<1440, 256, 0, stream>>>(sv, sgruswz, sg_f_bih, sg_b_bih, gx);
  sgru_seq_k<<<2, 256, 0, stream>>>(sgruswz, sg_f_bhh, sg_b_bhh, gx, doc);
  tail_k<<<1, 768, 0, stream>>>(doc, user_feats,
                                rfc_w1, rfc_b1, rfc_w2, rfc_b2,
                                pfc_w1, pfc_b1, pfc_w2, pfc_b2,
                                rg_f_wih, rg_f_whh, rg_f_bih, rg_f_bhh,
                                rg_b_wih, rg_b_whh, rg_b_bih, rg_b_bhh,
                                (float*)d_out);
}
